// PhysicsGuidedBlockPatchLevel_79834852098105
// MI455X (gfx1250) — compile-verified, matrix operand order checked by reading, not run on hardware
//
#include <hip/hip_runtime.h>
#include <hip/hip_bf16.h>
#include <math.h>

#define DIM      768
#define HEADS    12
#define HEAD_DIM 64
#define HIDDEN   3072
#define SEQ      2048
#define BATCH    2
#define NTOK     (BATCH*SEQ)
#define ATTN_SCALE 0.125f   /* 64^-0.5 */
#define LN_EPS   1e-5f

// LDS row pitches chosen for conflict-free ds_load_b128 fragment reads
// (64 banks x 4B): pitch 40 halves = 20 banks, pitch 72 halves = 36 banks.
#define GPITCH 40
#define APITCH 72

typedef __attribute__((ext_vector_type(16))) _Float16 v16h;
typedef __attribute__((ext_vector_type(8)))  _Float16 v8h;
typedef __attribute__((ext_vector_type(8)))  float    v8f;

union V16H { v16h v; v8h h[2]; };

// 16x32 f16 WMMA fragment: halves j=0..7 at p[0..7] (k=hi*8+0..7),
// j=8..15 at p[16..23] (k=16+hi*8+0..7). p = &tile[row*pitch + hi*8].
static __device__ inline v16h load_frag(const _Float16* p) {
    V16H f;
    f.h[0] = *(const v8h*)(p);
    f.h[1] = *(const v8h*)(p + 16);
    return f.v;
}

// ---- CDNA5 async global->LDS copy (ASYNCcnt-tracked), via inline asm ------
// LDS destination address comes from the vdst VGPR; generic shared pointers carry the LDS byte
// offset in their low 32 bits (flat->LDS truncation per ISA 10.2).
static __device__ inline void async_cp16(const _Float16* g, _Float16* lds) {
    unsigned loff = (unsigned)(size_t)lds;
    asm volatile("global_load_async_to_lds_b128 %0, %1, off"
                 :: "v"(loff), "v"(g) : "memory");
}
static __device__ inline void wait_async0() {
    asm volatile("s_wait_asynccnt 0x0" ::: "memory");
}
static __device__ inline void wait_async3() {
    asm volatile("s_wait_asynccnt 0x3" ::: "memory");
}

// ---------------------------------------------------------------------------
// Weight convert + transpose: W[K][Nw] f32 -> WT[Nw][K] f16.
// ---------------------------------------------------------------------------
__global__ void w_transpose_f16(const float* __restrict__ W,
                                _Float16* __restrict__ WT, int K, int Nw) {
    size_t idx = (size_t)blockIdx.x * 256 + threadIdx.x;
    if (idx >= (size_t)K * Nw) return;
    int n = (int)(idx / K), k = (int)(idx % K);
    WT[idx] = (_Float16)W[(size_t)k * Nw + n];
}

// ---------------------------------------------------------------------------
// LayerNorm one row of 768 f32 -> f16. Block = 256 threads.
// ---------------------------------------------------------------------------
__global__ __launch_bounds__(256) void layernorm_f16(
    const float* __restrict__ x, const float* __restrict__ g,
    const float* __restrict__ b, _Float16* __restrict__ out) {
    __shared__ float rs[256], rq[256];
    const int row = blockIdx.x, tid = threadIdx.x;
    const float* xr = x + (size_t)row * DIM;
    float v0 = xr[tid], v1 = xr[tid + 256], v2 = xr[tid + 512];
    rs[tid] = v0 + v1 + v2;
    rq[tid] = v0*v0 + v1*v1 + v2*v2;
    __syncthreads();
    for (int s = 128; s > 0; s >>= 1) {
        if (tid < s) { rs[tid] += rs[tid + s]; rq[tid] += rq[tid + s]; }
        __syncthreads();
    }
    float mean = rs[0] * (1.0f / DIM);
    float var  = rq[0] * (1.0f / DIM) - mean * mean;
    float inv  = rsqrtf(var + LN_EPS);
    _Float16* orow = out + (size_t)row * DIM;
    orow[tid]       = (_Float16)((v0 - mean) * inv * g[tid]       + b[tid]);
    orow[tid + 256] = (_Float16)((v1 - mean) * inv * g[tid + 256] + b[tid + 256]);
    orow[tid + 512] = (_Float16)((v2 - mean) * inv * g[tid + 512] + b[tid + 512]);
}

// ---------------------------------------------------------------------------
// WMMA GEMM: C[M][N] = A[M][K](f16) @ BT[N][K](f16)^T + bias, epilogues:
//   EPI 0: + bias                  -> f16 out
//   EPI 1: + bias + residual(f32)  -> f32 out
//   EPI 2: + bias, exact GELU      -> f16 out
// Block 256 thr = 8 waves; tile 128x64; K-step 32; async double-buffered LDS.
// Each wave issues exactly 3 async copies per buffer (2 A-chunks + 1 B-chunk),
// so s_wait_asynccnt 3 after issuing the next buffer == current buffer done.
// ---------------------------------------------------------------------------
template<int EPI>
__global__ __launch_bounds__(256) void gemm_wmma_epi(
    const _Float16* __restrict__ A, int lda,
    const _Float16* __restrict__ BT, int ldb,
    const float* __restrict__ bias,
    const float* __restrict__ residual,
    float* __restrict__ outF, _Float16* __restrict__ outH, int ldo,
    int K) {
    __shared__ __align__(16) _Float16 As[2][128 * GPITCH];
    __shared__ __align__(16) _Float16 Bs[2][64 * GPITCH];
    const int tid  = threadIdx.x;
    const int wave = tid >> 5, lane = tid & 31;
    const int ln = lane & 15, hi = lane >> 4;
    const int rowBase = blockIdx.y * 128, colBase = blockIdx.x * 64;

    // per-thread staging coordinates
    const int ar0 = (tid * 2) >> 2,       ac0 = ((tid * 2) & 3) * 8;
    const int ar1 = (tid * 2 + 1) >> 2,   ac1 = ((tid * 2 + 1) & 3) * 8;
    const int br  = tid >> 2,             bc  = (tid & 3) * 8;

    auto issue = [&](int buf, int k0) {
        async_cp16(A  + (size_t)(rowBase + ar0) * lda + k0 + ac0, &As[buf][ar0 * GPITCH + ac0]);
        async_cp16(A  + (size_t)(rowBase + ar1) * lda + k0 + ac1, &As[buf][ar1 * GPITCH + ac1]);
        async_cp16(BT + (size_t)(colBase + br ) * ldb + k0 + bc,  &Bs[buf][br  * GPITCH + bc ]);
    };

    v8f acc[4] = {};
    const int nk = K / 32;
    issue(0, 0);
    for (int i = 0; i < nk; ++i) {
        const int cur = i & 1;
        if (i + 1 < nk) { issue(cur ^ 1, (i + 1) * 32); wait_async3(); }
        else            { wait_async0(); }
        __syncthreads();
        v16h a = load_frag(&As[cur][(wave * 16 + ln) * GPITCH + hi * 8]);
#pragma unroll
        for (int t = 0; t < 4; ++t) {
            v16h bf = load_frag(&Bs[cur][(t * 16 + ln) * GPITCH + hi * 8]);
            acc[t] = __builtin_amdgcn_wmma_f32_16x16x32_f16(
                false, a, false, bf, (short)0, acc[t], false, false);
        }
        __syncthreads();   // all waves done reading `cur` before it is refilled
    }

#pragma unroll
    for (int t = 0; t < 4; ++t) {
        int col = colBase + t * 16 + ln;
        float bv = bias[col];
#pragma unroll
        for (int r = 0; r < 8; ++r) {
            int row = rowBase + wave * 16 + r + 8 * hi;
            float v = acc[t][r] + bv;
            if constexpr (EPI == 1) {
                v += residual[(size_t)row * ldo + col];
                outF[(size_t)row * ldo + col] = v;
            } else if constexpr (EPI == 2) {
                v = 0.5f * v * (1.0f + erff(v * 0.70710678118654752f));
                outH[(size_t)row * ldo + col] = (_Float16)v;
            } else {
                outH[(size_t)row * ldo + col] = (_Float16)v;
            }
        }
    }
}

// ---------------------------------------------------------------------------
// Fused flash-attention with elevation mask (reference semantics):
//   with u = e^{S-m}*mask, Z = sum e^{S-m}:  out = (sum u*V) / (sum u + 1e-8*Z)
// Block = 128 threads (4 waves) handles 64 queries of one (b,h).
// ---------------------------------------------------------------------------
__global__ __launch_bounds__(128) void attn_flash(
    const _Float16* __restrict__ qkv, const float* __restrict__ elev,
    const float* __restrict__ barrier_p, _Float16* __restrict__ attn) {
    __shared__ __align__(16) _Float16 Qs[64 * APITCH];
    __shared__ __align__(16) _Float16 Ks[64 * APITCH];
    __shared__ __align__(16) _Float16 Vt[64 * APITCH];     // [dim][key]
    __shared__ __align__(16) _Float16 Ps[4][16 * APITCH];  // per-wave P tile
    __shared__ float ke[64];
    const int tid  = threadIdx.x;
    const int wave = tid >> 5, lane = tid & 31;
    const int ln = lane & 15, hi = lane >> 4;
    const int b = blockIdx.z, h = blockIdx.y, q0 = blockIdx.x * 64;
    const size_t tb = (size_t)b * SEQ;
    const size_t ldq = 3 * DIM;
    const float bar = barrier_p[0];

    // Q tile 64x64
#pragma unroll
    for (int c = 0; c < 4; ++c) {
        int ci = tid + c * 128;
        int r = ci >> 3, col = (ci & 7) * 8;
        *(v8h*)&Qs[r * APITCH + col] =
            *(const v8h*)(qkv + (tb + q0 + r) * ldq + h * HEAD_DIM + col);
    }
    float ei[8];
#pragma unroll
    for (int r = 0; r < 8; ++r)
        ei[r] = elev[tb + q0 + wave * 16 + r + 8 * hi];

    float mrun[8], se[8], su[8];
#pragma unroll
    for (int r = 0; r < 8; ++r) { mrun[r] = -3.4e38f; se[r] = 0.f; su[r] = 0.f; }
    v8f oacc[4] = {};

    for (int kb = 0; kb < SEQ; kb += 64) {
        // Stage K tile row-major [key][d] and V transposed [d][key].
#pragma unroll
        for (int c = 0; c < 4; ++c) {
            int ci = tid + c * 128;
            int r = ci >> 3, col = (ci & 7) * 8;
            *(v8h*)&Ks[r * APITCH + col] =
                *(const v8h*)(qkv + (tb + kb + r) * ldq + DIM + h * HEAD_DIM + col);
            v8h vv = *(const v8h*)(qkv + (tb + kb + r) * ldq + 2 * DIM + h * HEAD_DIM + col);
#pragma unroll
            for (int e = 0; e < 8; ++e) Vt[(col + e) * APITCH + r] = vv[e];
        }
        if (tid < 64) ke[tid] = elev[tb + kb + tid];
        // Prefetch next block's K/V rows into L2/WGP$ while we compute.
        if (kb + 64 < SEQ) {
            int pr = tid & 63;
            const _Float16* pb = qkv + (tb + kb + 64 + pr) * ldq +
                                 ((tid & 64) ? 2 * DIM : DIM) + h * HEAD_DIM;
            __builtin_prefetch(pb, 0, 0);
        }
        __syncthreads();

        // S = Q @ K^T  (16x64 per wave, 8 WMMAs)
        v8f s[4] = {};
#pragma unroll
        for (int kk = 0; kk < 64; kk += 32) {
            v16h a = load_frag(&Qs[(wave * 16 + ln) * APITCH + kk + hi * 8]);
#pragma unroll
            for (int t = 0; t < 4; ++t) {
                v16h bf = load_frag(&Ks[(t * 16 + ln) * APITCH + kk + hi * 8]);
                s[t] = __builtin_amdgcn_wmma_f32_16x16x32_f16(
                    false, a, false, bf, (short)0, s[t], false, false);
            }
        }
#pragma unroll
        for (int t = 0; t < 4; ++t) s[t] *= ATTN_SCALE;

        // Online row max (row m = r + 8*hi spans the 16 lanes of this half).
        float mnew[8], alpha[8];
#pragma unroll
        for (int r = 0; r < 8; ++r) {
            float mx = fmaxf(fmaxf(s[0][r], s[1][r]), fmaxf(s[2][r], s[3][r]));
#pragma unroll
            for (int off = 8; off >= 1; off >>= 1)
                mx = fmaxf(mx, __shfl_xor(mx, off, 32));
            mnew[r]  = fmaxf(mrun[r], mx);
            alpha[r] = __expf(mrun[r] - mnew[r]);
        }

        float es[8], us[8];
#pragma unroll
        for (int r = 0; r < 8; ++r) { es[r] = 0.f; us[r] = 0.f; }
#pragma unroll
        for (int t = 0; t < 4; ++t) {
            float kev = ke[t * 16 + ln];
#pragma unroll
            for (int r = 0; r < 8; ++r) {
                float p   = __expf(s[t][r] - mnew[r]);
                float msk = 1.0f / (1.0f + __expf(bar * (kev - ei[r])));
                msk = fminf(fmaxf(msk, 1e-6f), 1.0f);
                float u = p * msk;
                es[r] += p; us[r] += u;
                Ps[wave][(r + 8 * hi) * APITCH + t * 16 + ln] = (_Float16)u;
            }
        }
#pragma unroll
        for (int r = 0; r < 8; ++r) {
            float e = es[r], uu = us[r];
#pragma unroll
            for (int off = 8; off >= 1; off >>= 1) {
                e  += __shfl_xor(e,  off, 32);
                uu += __shfl_xor(uu, off, 32);
            }
            se[r] = se[r] * alpha[r] + e;
            su[r] = su[r] * alpha[r] + uu;
#pragma unroll
            for (int t = 0; t < 4; ++t) oacc[t][r] *= alpha[r];
            mrun[r] = mnew[r];
        }

        // O += P @ V (P from per-wave LDS; same-wave DS ops are in-order).
#pragma unroll
        for (int kk = 0; kk < 64; kk += 32) {
            v16h a = load_frag(&Ps[wave][ln * APITCH + kk + hi * 8]);
#pragma unroll
            for (int t = 0; t < 4; ++t) {
                v16h bf = load_frag(&Vt[(t * 16 + ln) * APITCH + kk + hi * 8]);
                oacc[t] = __builtin_amdgcn_wmma_f32_16x16x32_f16(
                    false, a, false, bf, (short)0, oacc[t], false, false);
            }
        }
        __syncthreads();
    }

#pragma unroll
    for (int t = 0; t < 4; ++t)
#pragma unroll
        for (int r = 0; r < 8; ++r) {
            float denom = su[r] + 1e-8f * se[r];
            float v = oacc[t][r] / denom;
            size_t row = tb + q0 + wave * 16 + r + 8 * hi;
            attn[row * DIM + h * HEAD_DIM + t * 16 + ln] = (_Float16)v;
        }
}

// ---------------------------------------------------------------------------
extern "C" void kernel_launch(void* const* d_in, const int* in_sizes, int n_in,
                              void* d_out, int out_size, void* d_ws, size_t ws_size,
                              hipStream_t stream) {
    const float* x       = (const float*)d_in[0];
    const float* elev    = (const float*)d_in[1];
    const float* qkv_w   = (const float*)d_in[2];
    const float* qkv_bi  = (const float*)d_in[3];
    const float* proj_w  = (const float*)d_in[4];
    const float* proj_bi = (const float*)d_in[5];
    const float* ln1_g   = (const float*)d_in[6];
    const float* ln1_b   = (const float*)d_in[7];
    const float* ln2_g   = (const float*)d_in[8];
    const float* ln2_b   = (const float*)d_in[9];
    const float* fc1_w   = (const float*)d_in[10];
    const float* fc1_bi  = (const float*)d_in[11];
    const float* fc2_w   = (const float*)d_in[12];
    const float* fc2_bi  = (const float*)d_in[13];
    const float* barrier_s = (const float*)d_in[14];
    float* out = (float*)d_out;
    (void)in_sizes; (void)n_in; (void)out_size; (void)ws_size;

    char* ws = (char*)d_ws;
    size_t off = 0;
    auto alloc = [&](size_t bytes) {
        size_t o = off; off += (bytes + 255) & ~(size_t)255; return o;
    };

    const size_t M = NTOK;
    const size_t h1_bytes  = M * DIM * 2;
    const size_t qkv_bytes = M * 3 * DIM * 2;
    const size_t g_bytes   = M * HIDDEN * 2;
    // region0: h1 + qkv while attention is live; reused for MLP hidden after.
    size_t reg0 = alloc(h1_bytes + qkv_bytes > g_bytes ? h1_bytes + qkv_bytes : g_bytes);
    _Float16* h1    = (_Float16*)(ws + reg0);
    _Float16* qkvA  = (_Float16*)(ws + reg0 + h1_bytes);
    _Float16* gbuf  = (_Float16*)(ws + reg0);
    _Float16* attn  = (_Float16*)(ws + alloc(M * DIM * 2));
    float*    x1    = (float*)   (ws + alloc(M * DIM * 4));
    _Float16* h2    = (_Float16*)(ws + alloc(M * DIM * 2));
    _Float16* qkvWT = (_Float16*)(ws + alloc((size_t)DIM * 3 * DIM * 2));
    _Float16* projWT= (_Float16*)(ws + alloc((size_t)DIM * DIM * 2));
    _Float16* fc1WT = (_Float16*)(ws + alloc((size_t)DIM * HIDDEN * 2));
    _Float16* fc2WT = (_Float16*)(ws + alloc((size_t)HIDDEN * DIM * 2));

    auto tgrid = [](size_t n) { return dim3((unsigned)((n + 255) / 256)); };
    w_transpose_f16<<<tgrid((size_t)DIM * 3 * DIM), 256, 0, stream>>>(qkv_w, qkvWT, DIM, 3 * DIM);
    w_transpose_f16<<<tgrid((size_t)DIM * DIM),     256, 0, stream>>>(proj_w, projWT, DIM, DIM);
    w_transpose_f16<<<tgrid((size_t)DIM * HIDDEN),  256, 0, stream>>>(fc1_w, fc1WT, DIM, HIDDEN);
    w_transpose_f16<<<tgrid((size_t)HIDDEN * DIM),  256, 0, stream>>>(fc2_w, fc2WT, HIDDEN, DIM);

    layernorm_f16<<<dim3((unsigned)M), 256, 0, stream>>>(x, ln1_g, ln1_b, h1);

    gemm_wmma_epi<0><<<dim3(3 * DIM / 64, M / 128), 256, 0, stream>>>(
        h1, DIM, qkvWT, DIM, qkv_bi, nullptr, nullptr, qkvA, 3 * DIM, DIM);

    attn_flash<<<dim3(SEQ / 64, HEADS, BATCH), 128, 0, stream>>>(qkvA, elev, barrier_s, attn);

    gemm_wmma_epi<1><<<dim3(DIM / 64, M / 128), 256, 0, stream>>>(
        attn, DIM, projWT, DIM, proj_bi, x, x1, nullptr, DIM, DIM);

    layernorm_f16<<<dim3((unsigned)M), 256, 0, stream>>>(x1, ln2_g, ln2_b, h2);

    gemm_wmma_epi<2><<<dim3(HIDDEN / 64, M / 128), 256, 0, stream>>>(
        h2, DIM, fc1WT, DIM, fc1_bi, nullptr, nullptr, gbuf, HIDDEN, DIM);

    gemm_wmma_epi<1><<<dim3(DIM / 64, M / 128), 256, 0, stream>>>(
        gbuf, HIDDEN, fc2WT, HIDDEN, fc2_bi, x1, out, nullptr, DIM, HIDDEN);
}
